// VSSMBlock_Mamba2_12086037971085
// MI455X (gfx1250) — matrix operand order repaired and checked by reading, not compiled, not run on hardware
//
#include <hip/hip_runtime.h>
#include <hip/hip_bf16.h>

// ---------------------------------------------------------------------------
// Mamba2 VSSM block for gfx1250 (MI455X).  All matmul-shaped einsums except k_states run on
// v_wmma_f32_16x16x32_bf16 (fp32 accumulate).  GEMM B-tiles are staged into
// LDS with the CDNA5 async copy path (ASYNCcnt) via inline asm.
// Requires ~450 MB of d_ws.
// ---------------------------------------------------------------------------

#define TOKS   32768L          // B*L = 8*4096
#define DINNER 512
#define CONVD  640
#define NHEADS 8
#define NPROJ  1160
#define NPAD1  1216            // 1160 padded to 64-multiple

#define USE_ASYNC_LDS 1        // flip to 0 if the assembler rejects the mnemonic

typedef __attribute__((ext_vector_type(16))) __bf16 v16bf;
typedef __attribute__((ext_vector_type(8)))  float  v8f;

union Frag { uint4 q[2]; v16bf v; unsigned short s[16]; };

__device__ __forceinline__ unsigned short f2bf(float f) {
  unsigned u = __float_as_uint(f);
  u += 0x7FFFu + ((u >> 16) & 1u);          // round-to-nearest-even
  return (unsigned short)(u >> 16);
}
__device__ __forceinline__ float bf2f(unsigned short h) {
  return __uint_as_float(((unsigned)h) << 16);
}

// Async global->LDS 16-byte copy per lane (VGLOBAL async op, ASYNCcnt).
__device__ __forceinline__ void copy16_to_lds(const unsigned short* gsrc,
                                              unsigned short* ldst) {
#if USE_ASYNC_LDS
  unsigned lds_off = (unsigned)(unsigned long long)
      (__attribute__((address_space(3))) const unsigned short*)ldst;
  unsigned long long ga = (unsigned long long)gsrc;
  asm volatile("global_load_async_to_lds_b128 %0, %1, off"
               :
               : "v"(lds_off), "v"(ga)
               : "memory");
#else
  *(uint4*)ldst = *(const uint4*)gsrc;
#endif
}

__device__ __forceinline__ void wait_async_lds() {
#if USE_ASYNC_LDS
  asm volatile("s_wait_asynccnt 0" ::: "memory");
#endif
}

// A fragment (16x32 bf16): lane = row (m0 + lane&15), K split by lane half.
__device__ __forceinline__ v16bf load_a_frag(const unsigned short* src, int ld,
                                             int m0, int k0, int lane) {
  int row = m0 + (lane & 15);
  int kb  = k0 + ((lane >> 4) << 3);
  const unsigned short* p = src + (size_t)row * ld + kb;
  Frag f;
  f.q[0] = *(const uint4*)p;          // K = kb .. kb+7
  f.q[1] = *(const uint4*)(p + 16);   // K = kb+16 .. kb+23
  return f.v;
}

// Same, with a per-row fp32 scale factor exp(acs[row]) folded in (for Y_off).
__device__ __forceinline__ v16bf load_a_frag_scaled(const unsigned short* src, int ld,
                                                    int m0, int k0, int lane,
                                                    const float* acs) {
  int row = m0 + (lane & 15);
  int kb  = k0 + ((lane >> 4) << 3);
  const unsigned short* p = src + (size_t)row * ld + kb;
  float sc = __expf(acs[row]);
  Frag f;
  f.q[0] = *(const uint4*)p;
  f.q[1] = *(const uint4*)(p + 16);
#pragma unroll
  for (int j = 0; j < 16; ++j) f.s[j] = f2bf(bf2f(f.s[j]) * sc);
  return f.v;
}

// B fragment (32x16 bf16): lane = column (n0 + lane&15), K split by lane half like A.
__device__ __forceinline__ v16bf load_b_frag(const unsigned short* src, int ld,
                                             int k0, int n0, int lane) {
  int kb = k0 + ((lane >> 4) << 3);
  const unsigned short* p = src + (size_t)kb * ld + n0 + (lane & 15);
  Frag f;
#pragma unroll
  for (int j = 0; j < 8; ++j) {
    f.s[j]     = p[(size_t)j * ld];
    f.s[8 + j] = p[(size_t)(16 + j) * ld];
  }
  return f.v;
}

__device__ __forceinline__ v8f wmma_bf16(v16bf a, v16bf b, v8f c) {
  return __builtin_amdgcn_wmma_f32_16x16x32_bf16(false, a, false, b,
                                                 (short)0, c, false, false);
}

// ---------------------------------------------------------------------------
// K0: LayerNorm over channels -> token-major bf16 xn[TOKS][256]
// ---------------------------------------------------------------------------
__global__ __launch_bounds__(256) void k_layernorm(const float* __restrict__ x,
                                                   const float* __restrict__ w,
                                                   const float* __restrict__ bia,
                                                   unsigned short* __restrict__ xn) {
  int warp = threadIdx.x >> 5, lane = threadIdx.x & 31;
  long t  = (long)blockIdx.x * 8 + warp;
  long b  = t >> 12, hw = t & 4095;
  const float* base = x + b * (256L * 4096L) + hw;
  float vals[8], s = 0.f, sq = 0.f;
#pragma unroll
  for (int i = 0; i < 8; ++i) {
    float v = base[(long)(lane + 32 * i) * 4096L];
    vals[i] = v; s += v; sq += v * v;
  }
#pragma unroll
  for (int m = 16; m > 0; m >>= 1) {
    s  += __shfl_xor(s,  m, 32);
    sq += __shfl_xor(sq, m, 32);
  }
  float mu  = s * (1.f / 256.f);
  float var = sq * (1.f / 256.f) - mu * mu;
  float inv = rsqrtf(var + 1e-5f);
#pragma unroll
  for (int i = 0; i < 8; ++i) {
    int c = lane + 32 * i;
    xn[t * 256 + c] = f2bf((vals[i] - mu) * inv * w[c] + bia[c]);
  }
}

// ---------------------------------------------------------------------------
// Weight prep: transpose + cast to bf16 (K x Npad row-major, zero padded)
// ---------------------------------------------------------------------------
__global__ void k_prep_w1(const float* __restrict__ W, unsigned short* __restrict__ Wt) {
  long idx = (long)blockIdx.x * 256 + threadIdx.x;           // k*NPAD1 + n
  if (idx >= 256L * NPAD1) return;
  int n = (int)(idx % NPAD1), k = (int)(idx / NPAD1);
  Wt[idx] = (n < NPROJ) ? f2bf(W[(long)n * 256 + k]) : (unsigned short)0;
}
__global__ void k_prep_w2(const float* __restrict__ W, unsigned short* __restrict__ Wt) {
  long idx = (long)blockIdx.x * 256 + threadIdx.x;           // k*256 + n
  if (idx >= 512L * 256L) return;
  int n = (int)(idx & 255), k = (int)(idx >> 8);
  Wt[idx] = f2bf(W[(long)n * 512 + k]);
}

// ---------------------------------------------------------------------------
// Tiled bf16 WMMA GEMM: C[M x ldc](f32) = A[M x K](bf16) * Bt[K x ldb](bf16)
// block = 8 waves, each wave a 16x64 strip of a 128x64 block tile.
// B tile (32x64) is double-buffered in LDS via async copy (ASYNCcnt path).
// grid = (Npad/64, M/128)
// ---------------------------------------------------------------------------
__global__ __launch_bounds__(256) void k_gemm_bf16(const unsigned short* __restrict__ A,
                                                   const unsigned short* __restrict__ Bt,
                                                   float* __restrict__ C,
                                                   int M, int N, int K,
                                                   int lda, int ldb, int ldc) {
  __shared__ __align__(16) unsigned short Bl[2][32][72];   // padded pitch vs banks
  int tid  = threadIdx.x;
  int wave = tid >> 5, lane = tid & 31;
  int m0 = blockIdx.y * 128 + wave * 16;
  int n0 = blockIdx.x * 64;

  int srow = tid >> 3;                 // 0..31   (K row within tile)
  int scol = (tid & 7) * 8;            // 0..56   (8 bf16 = 16 B per thread)

  v8f vz = {0.f,0.f,0.f,0.f,0.f,0.f,0.f,0.f};
  v8f acc[4] = {vz, vz, vz, vz};

  int nk = K / 32;
  // stage k-step 0 into buffer 0
  copy16_to_lds(Bt + (size_t)srow * ldb + n0 + scol, &Bl[0][srow][scol]);

  for (int kt = 0; kt < nk; ++kt) {
    int cur = kt & 1;
    int k0  = kt * 32;
    wait_async_lds();
    __syncthreads();                               // buffer `cur` ready everywhere
    if (kt + 1 < nk)                               // stage next tile while computing
      copy16_to_lds(Bt + (size_t)(k0 + 32 + srow) * ldb + n0 + scol,
                    &Bl[cur ^ 1][srow][scol]);
    if (k0 + 64 < K)
      __builtin_prefetch(A + (size_t)m0 * lda + k0 + 64, 0, 1); // global_prefetch_b8

    v16bf a = load_a_frag(A, lda, m0, k0, lane);
#pragma unroll
    for (int t = 0; t < 4; ++t) {
      v16bf b = load_b_frag(&Bl[cur][0][0], 72, 0, 16 * t, lane);
      acc[t] = wmma_bf16(a, b, acc[t]);
    }
    __syncthreads();                               // done reading `cur` before reuse
  }
#pragma unroll
  for (int t = 0; t < 4; ++t)
#pragma unroll
    for (int r = 0; r < 8; ++r) {
      int m = m0 + r + 8 * (lane >> 4);
      int n = n0 + 16 * t + (lane & 15);
      if (n < N) C[(size_t)m * ldc + n] = acc[t][r];
    }
}

// ---------------------------------------------------------------------------
// K2b: dt softplus + dA = -softplus(dt+bias)*exp(A_log)
// ---------------------------------------------------------------------------
__global__ void k_dt(const float* __restrict__ zx, const float* __restrict__ dt_bias,
                     const float* __restrict__ A_log,
                     float* __restrict__ dt_sp, float* __restrict__ dA) {
  long idx = (long)blockIdx.x * 256 + threadIdx.x;            // t*8 + h
  if (idx >= TOKS * NHEADS) return;
  int h = (int)(idx & 7); long t = idx >> 3;
  float v  = zx[t * NPAD1 + 1152 + h] + dt_bias[h];
  float sp = (v > 20.f) ? v : log1pf(__expf(v));
  dt_sp[idx] = sp;
  dA[idx]    = -sp * __expf(A_log[h]);
}

// ---------------------------------------------------------------------------
// K2: depthwise 3x3 conv (SAME) + bias + SiLU over xBC cols [512,1152) of zx.
// Emits bf16 xbcb[t][640] = [X*dt | B | C] and fp32 raw xs[t][512].
// ---------------------------------------------------------------------------
__global__ __launch_bounds__(256) void k_conv(const float* __restrict__ zx,
                                              const float* __restrict__ cw,
                                              const float* __restrict__ cb,
                                              const float* __restrict__ dt_sp,
                                              unsigned short* __restrict__ xbcb,
                                              float* __restrict__ xs) {
  long idx = (long)blockIdx.x * 256 + threadIdx.x;            // t*640 + j
  if (idx >= TOKS * CONVD) return;
  int  j = (int)(idx % CONVD);
  long t = idx / CONVD;
  int  b = (int)(t >> 12), hw = (int)(t & 4095);
  int  hh = hw >> 6, ww = hw & 63;
  float acc = cb[j];
#pragma unroll
  for (int dy = 0; dy < 3; ++dy)
#pragma unroll
    for (int dx = 0; dx < 3; ++dx) {
      int yy = hh + dy - 1, xx = ww + dx - 1;
      if (yy >= 0 && yy < 64 && xx >= 0 && xx < 64)
        acc += cw[j * 9 + dy * 3 + dx] *
               zx[((long)b * 4096 + yy * 64 + xx) * NPAD1 + 512 + j];
    }
  float v = acc / (1.f + __expf(-acc));                       // SiLU
  if (j < DINNER) {
    xs[t * DINNER + j] = v;
    xbcb[t * CONVD + j] = f2bf(v * dt_sp[t * 8 + (j >> 6)]);  // X*dt (matmul input)
  } else {
    xbcb[t * CONVD + j] = f2bf(v);                            // B / C
  }
}

// ---------------------------------------------------------------------------
// K3: per-(b,h,chunk) inclusive cumsum of dA -> Acs, chunk totals -> Asum
// ---------------------------------------------------------------------------
__global__ __launch_bounds__(256) void k_cumsum(const float* __restrict__ dA,
                                                float* __restrict__ Acs,
                                                float* __restrict__ Asum) {
  int blk = blockIdx.x;                       // (b*8+h)*16 + ch
  int ch = blk & 15, bh = blk >> 4, h = bh & 7, b = bh >> 3;
  int q = threadIdx.x;
  long t = (long)b * 4096 + (long)ch * 256 + q;
  __shared__ float buf[256];
  buf[q] = dA[t * 8 + h];
  __syncthreads();
  for (int off = 1; off < 256; off <<= 1) {
    float v = (q >= off) ? buf[q - off] : 0.f;
    __syncthreads();
    buf[q] += v;
    __syncthreads();
  }
  Acs[(long)blk * 256 + q] = buf[q];
  if (q == 255) Asum[blk] = buf[255];
}

// ---------------------------------------------------------------------------
// K_states: states[b,ch,h][p][n] = sum_q X[q][p]*exp(Asum-Acs[q])*B[q][n]
// X*decay staged in LDS (bf16); B rows streamed from L2.
// blockIdx = ((b*16+ch)*8 + h)
// ---------------------------------------------------------------------------
__global__ __launch_bounds__(256) void k_states(const unsigned short* __restrict__ xbcb,
                                                const float* __restrict__ Acs,
                                                const float* __restrict__ Asum,
                                                float* __restrict__ states) {
  int blk = blockIdx.x;
  int h = blk & 7, bc = blk >> 3, ch = bc & 15, b = bc >> 4;
  long tokbase = (long)b * 4096 + (long)ch * 256;
  int  acsblk  = (b * 8 + h) * 16 + ch;
  float asum = Asum[acsblk];
  __shared__ __align__(16) unsigned short Xl[256 * 64];       // 32 KB
  int tid = threadIdx.x;
  for (int i = tid; i < 256 * 64 / 8; i += 256) {
    int q = i >> 3, p = (i & 7) * 8;
    Frag f;
    f.q[0] = *(const uint4*)(xbcb + (tokbase + q) * CONVD + h * 64 + p);
    float d = __expf(asum - Acs[(long)acsblk * 256 + q]);
#pragma unroll
    for (int e = 0; e < 8; ++e) f.s[e] = f2bf(bf2f(f.s[e]) * d);
    *(uint4*)(Xl + q * 64 + p) = f.q[0];
  }
  __syncthreads();
  int p = tid >> 2, nb = (tid & 3) * 16;
  float acc[16];
#pragma unroll
  for (int e = 0; e < 16; ++e) acc[e] = 0.f;
  for (int q = 0; q < 256; ++q) {
    float xv = bf2f(Xl[q * 64 + p]);
    Frag f;
    f.q[0] = *(const uint4*)(xbcb + (tokbase + q) * CONVD + 512 + nb);
    f.q[1] = *(const uint4*)(xbcb + (tokbase + q) * CONVD + 512 + nb + 8);
#pragma unroll
    for (int e = 0; e < 16; ++e) acc[e] += xv * bf2f(f.s[e]);
  }
  float* dst = states + (long)blk * 4096 + p * 64 + nb;
#pragma unroll
  for (int e = 0; e < 16; ++e) dst[e] = acc[e];
}

// ---------------------------------------------------------------------------
// K4: sequential inter-chunk scan (16 steps).  prevT (n x p, bf16) per chunk.
// blockIdx = b*8 + h
// ---------------------------------------------------------------------------
__global__ __launch_bounds__(256) void k_scan(const float* __restrict__ states,
                                              const float* __restrict__ Asum,
                                              unsigned short* __restrict__ prevT) {
  int bh = blockIdx.x;
  int h = bh & 7, b = bh >> 3;
  int tid = threadIdx.x;
  float P[16];
#pragma unroll
  for (int j = 0; j < 16; ++j) P[j] = 0.f;
  for (int z = 0; z < 16; ++z) {
    long sblk = ((long)(b * 16 + z) * 8 + h) * 4096;
    float ez = __expf(Asum[(b * 8 + h) * 16 + z]);
#pragma unroll
    for (int j = 0; j < 16; ++j) {
      int o = tid + 256 * j;
      int p = o >> 6, n = o & 63;
      prevT[sblk + n * 64 + p] = f2bf(P[j]);        // prev state for chunk z
      P[j] = P[j] * ez + states[sblk + p * 64 + n];
    }
  }
}

// ---------------------------------------------------------------------------
// K5: SSD core per (b, chunk, head): Y = tril-decay(C.B^T).X + exp(Acs).C.prevT
// All three matmuls on WMMA bf16.  blockIdx = ((b*16+ch)*8 + h)
// ---------------------------------------------------------------------------
__global__ __launch_bounds__(256) void k_ssd(const unsigned short* __restrict__ xbcb,
                                             const unsigned short* __restrict__ prevT,
                                             const float* __restrict__ Acs,
                                             float* __restrict__ y_ssd) {
  int blk = blockIdx.x;
  int h = blk & 7, bc = blk >> 3, ch = bc & 15, b = bc >> 4;
  long tokbase = (long)b * 4096 + (long)ch * 256;
  int  acsblk  = (b * 8 + h) * 16 + ch;

  __shared__ __align__(16) unsigned short BlT[64][272];   // B^T (n x s), ~34.8 KB
  __shared__ __align__(16) unsigned short Sl[8][32][40];  // per-wave masked scores
  __shared__ float AcsL[256];

  int tid = threadIdx.x, lane = tid & 31, w = tid >> 5;

  for (int i = tid; i < 64 * 256; i += 256) {
    int n = i & 63, s = i >> 6;
    BlT[n][s] = xbcb[(tokbase + s) * CONVD + 512 + n];
  }
  AcsL[tid & 255] = Acs[(long)acsblk * 256 + (tid & 255)];
  __syncthreads();

  const unsigned short* Cbase = xbcb + tokbase * CONVD + 576;    // ld CONVD
  const unsigned short* Xbase = xbcb + tokbase * CONVD + h * 64; // ld CONVD
  const unsigned short* Pbase = prevT + (long)blk * 4096;        // ld 64

  int q0 = w * 32;
  v8f vz = {0.f,0.f,0.f,0.f,0.f,0.f,0.f,0.f};
  v8f yacc[2][4];
#pragma unroll
  for (int i = 0; i < 2; ++i)
#pragma unroll
    for (int j = 0; j < 4; ++j) yacc[i][j] = vz;

  for (int sb = 0; sb < 8; ++sb) {
    int s0 = sb * 32;
    v8f sacc[2][2] = {{vz, vz}, {vz, vz}};
#pragma unroll
    for (int kk = 0; kk < 64; kk += 32) {
      v16bf a0 = load_a_frag(Cbase, CONVD, q0,      kk, lane);
      v16bf a1 = load_a_frag(Cbase, CONVD, q0 + 16, kk, lane);
      v16bf b0 = load_b_frag(&BlT[0][0], 272, kk, s0,      lane);
      v16bf b1 = load_b_frag(&BlT[0][0], 272, kk, s0 + 16, lane);
      sacc[0][0] = wmma_bf16(a0, b0, sacc[0][0]);
      sacc[0][1] = wmma_bf16(a0, b1, sacc[0][1]);
      sacc[1][0] = wmma_bf16(a1, b0, sacc[1][0]);
      sacc[1][1] = wmma_bf16(a1, b1, sacc[1][1]);
    }
    // causal mask + segment decay, re-stage as bf16 A-fragments via LDS
#pragma unroll
    for (int i = 0; i < 2; ++i)
#pragma unroll
      for (int j = 0; j < 2; ++j)
#pragma unroll
        for (int r = 0; r < 8; ++r) {
          int q = q0 + 16 * i + r + 8 * (lane >> 4);
          int s = s0 + 16 * j + (lane & 15);
          float vv = (s <= q) ? sacc[i][j][r] * __expf(AcsL[q] - AcsL[s]) : 0.f;
          Sl[w][q - q0][s - s0] = f2bf(vv);
        }
    v16bf sa0 = load_a_frag(&Sl[w][0][0], 40, 0,  0, lane);
    v16bf sa1 = load_a_frag(&Sl[w][0][0], 40, 16, 0, lane);
#pragma unroll
    for (int j = 0; j < 4; ++j) {
      v16bf xb = load_b_frag(Xbase, CONVD, s0, 16 * j, lane);
      yacc[0][j] = wmma_bf16(sa0, xb, yacc[0][j]);
      yacc[1][j] = wmma_bf16(sa1, xb, yacc[1][j]);
    }
  }

  // Y_off: exp(Acs[q]) folded into C A-fragment, B = prevT (n x p)
#pragma unroll
  for (int kk = 0; kk < 64; kk += 32) {
    v16bf a0 = load_a_frag_scaled(Cbase, CONVD, q0,      kk, lane, AcsL);
    v16bf a1 = load_a_frag_scaled(Cbase, CONVD, q0 + 16, kk, lane, AcsL);
#pragma unroll
    for (int j = 0; j < 4; ++j) {
      v16bf pb = load_b_frag(Pbase, 64, kk, 16 * j, lane);
      yacc[0][j] = wmma_bf16(a0, pb, yacc[0][j]);
      yacc[1][j] = wmma_bf16(a1, pb, yacc[1][j]);
    }
  }

#pragma unroll
  for (int i = 0; i < 2; ++i)
#pragma unroll
    for (int j = 0; j < 4; ++j)
#pragma unroll
      for (int r = 0; r < 8; ++r) {
        int q = q0 + 16 * i + r + 8 * (lane >> 4);
        int p = 16 * j + (lane & 15);
        y_ssd[(tokbase + q) * DINNER + h * 64 + p] = yacc[i][j][r];
      }
}

// ---------------------------------------------------------------------------
// K6: y += D*xs; gate with silu(z); RMSNorm; emit bf16 for out_proj GEMM.
// one block per token
// ---------------------------------------------------------------------------
__global__ __launch_bounds__(256) void k_gate_rms(const float* __restrict__ y_ssd,
                                                  const float* __restrict__ xs,
                                                  const float* __restrict__ zx,
                                                  const float* __restrict__ Dv,
                                                  const float* __restrict__ rms_w,
                                                  unsigned short* __restrict__ ybf) {
  long t = blockIdx.x;
  int tid = threadIdx.x;
  __shared__ float red[256];
  float v[2], ss = 0.f;
#pragma unroll
  for (int k = 0; k < 2; ++k) {
    int d = tid + 256 * k;
    float ys = y_ssd[t * DINNER + d] + Dv[d >> 6] * xs[t * DINNER + d];
    float z  = zx[t * NPAD1 + d];
    float g  = z / (1.f + __expf(-z));
    v[k] = ys * g;
    ss += v[k] * v[k];
  }
  red[tid] = ss;
  __syncthreads();
  for (int off = 128; off > 0; off >>= 1) {
    if (tid < off) red[tid] += red[tid + off];
    __syncthreads();
  }
  float scale = rsqrtf(red[0] * (1.f / 512.f) + 1e-5f);
#pragma unroll
  for (int k = 0; k < 2; ++k) {
    int d = tid + 256 * k;
    ybf[t * DINNER + d] = f2bf(v[k] * scale * rms_w[d]);
  }
}

// ---------------------------------------------------------------------------
// K8: residual add back to NCHW
// ---------------------------------------------------------------------------
__global__ void k_final(const float* __restrict__ x, const float* __restrict__ out_t,
                        float* __restrict__ out) {
  long idx = (long)blockIdx.x * 256 + threadIdx.x;
  if (idx >= 8L * 256 * 4096) return;
  long b = idx >> 20;                 // 256*4096 = 2^20
  long rem = idx & 1048575;
  int c = (int)(rem >> 12), hw = (int)(rem & 4095);
  out[idx] = x[idx] + out_t[((long)b * 4096 + hw) * 256 + c];
}

// ---------------------------------------------------------------------------
extern "C" void kernel_launch(void* const* d_in, const int* in_sizes, int n_in,
                              void* d_out, int out_size, void* d_ws, size_t ws_size,
                              hipStream_t stream) {
  const float* x          = (const float*)d_in[0];
  const float* ln_w       = (const float*)d_in[1];
  const float* ln_b       = (const float*)d_in[2];
  const float* in_proj_w  = (const float*)d_in[3];
  const float* conv_w     = (const float*)d_in[4];
  const float* conv_b     = (const float*)d_in[5];
  const float* A_log      = (const float*)d_in[6];
  const float* Dv         = (const float*)d_in[7];
  const float* dt_bias    = (const float*)d_in[8];
  const float* rms_w      = (const float*)d_in[9];
  const float* out_proj_w = (const float*)d_in[10];
  float* out = (float*)d_out;

  char* ws = (char*)d_ws;
  size_t off = 0;
  auto take = [&](size_t bytes) -> char* {
    char* p = ws + off;
    off += (bytes + 255) & ~(size_t)255;
    return p;
  };
  unsigned short* xn     = (unsigned short*)take(TOKS * 256 * 2);        // 16.8 MB
  unsigned short* Wt1    = (unsigned short*)take(256L * NPAD1 * 2);      // 0.6 MB
  unsigned short* Wt2    = (unsigned short*)take(512L * 256 * 2);        // 0.26 MB
  float*          zx     = (float*)take(TOKS * NPAD1 * 4);               // 159 MB
  float*          dt_sp  = (float*)take(TOKS * NHEADS * 4);              // 1 MB
  float*          dA     = (float*)take(TOKS * NHEADS * 4);              // 1 MB
  unsigned short* xbcb   = (unsigned short*)take(TOKS * CONVD * 2);      // 42 MB
  float*          xs     = (float*)take(TOKS * DINNER * 4);              // 67 MB
  float*          Acs    = (float*)take(1024L * 256 * 4);                // 1 MB
  float*          Asum   = (float*)take(1024L * 4);
  float*          states = (float*)take(1024L * 4096 * 4);               // 16.8 MB
  unsigned short* prevT  = (unsigned short*)take(1024L * 4096 * 2);      // 8.4 MB
  float*          y_ssd  = (float*)take(TOKS * DINNER * 4);              // 67 MB
  unsigned short* ybf    = (unsigned short*)take(TOKS * DINNER * 2);     // 33.6 MB
  float*          out_t  = (float*)take(TOKS * 256 * 4);                 // 33.6 MB

  // weight prep
  k_prep_w1<<<(int)((256L * NPAD1 + 255) / 256), 256, 0, stream>>>(in_proj_w, Wt1);
  k_prep_w2<<<(int)((512L * 256 + 255) / 256), 256, 0, stream>>>(out_proj_w, Wt2);

  // layernorm -> bf16 tokens
  k_layernorm<<<TOKS / 8, 256, 0, stream>>>(x, ln_w, ln_b, xn);

  // in_proj GEMM: (32768 x 256) x (256 x 1216) -> zx (f32)
  k_gemm_bf16<<<dim3(NPAD1 / 64, TOKS / 128), 256, 0, stream>>>(
      xn, Wt1, zx, (int)TOKS, NPROJ, 256, 256, NPAD1, NPAD1);

  // dt softplus / dA, then depthwise conv + silu -> xbcb (bf16) + xs (f32)
  k_dt<<<(int)(TOKS * NHEADS / 256), 256, 0, stream>>>(zx, dt_bias, A_log, dt_sp, dA);
  k_conv<<<(int)(TOKS * CONVD / 256), 256, 0, stream>>>(zx, conv_w, conv_b, dt_sp, xbcb, xs);

  // SSD pipeline
  k_cumsum<<<1024, 256, 0, stream>>>(dA, Acs, Asum);
  k_states<<<1024, 256, 0, stream>>>(xbcb, Acs, Asum, states);
  k_scan<<<64, 256, 0, stream>>>(states, Asum, prevT);
  k_ssd<<<1024, 256, 0, stream>>>(xbcb, prevT, Acs, y_ssd);

  // gate + RMS norm -> bf16
  k_gate_rms<<<(int)TOKS, 256, 0, stream>>>(y_ssd, xs, zx, Dv, rms_w, ybf);

  // out_proj GEMM: (32768 x 512) x (512 x 256) -> out_t (f32)
  k_gemm_bf16<<<dim3(256 / 64, TOKS / 128), 256, 0, stream>>>(
      ybf, Wt2, out_t, (int)TOKS, 256, 512, 512, 256, 256);

  // residual add back to NCHW
  k_final<<<(int)(8L * 256 * 4096 / 256), 256, 0, stream>>>(x, out_t, out);
}
